// Seq2Seq_40321152975656
// MI455X (gfx1250) — compile-verified, matrix operand order checked by reading, not run on hardware
//
#include <hip/hip_runtime.h>
#include <hip/hip_bf16.h>
#include <math.h>

typedef __bf16 v16bf __attribute__((ext_vector_type(16)));
typedef __bf16 v8bf  __attribute__((ext_vector_type(8)));
typedef float  v8f   __attribute__((ext_vector_type(8)));
typedef int    v4i   __attribute__((ext_vector_type(4)));

#define GLOBAL_AS __attribute__((address_space(1)))
#define LDS_AS    __attribute__((address_space(3)))

#if defined(__has_builtin)
#if __has_builtin(__builtin_amdgcn_global_load_async_to_lds_b128)
#define HAVE_ASYNC_LDS 1
#endif
#endif
#ifndef HAVE_ASYNC_LDS
#define HAVE_ASYNC_LDS 0
#endif

static __device__ __forceinline__ v16bf cat8(v8bf lo, v8bf hi) {
  return __builtin_shufflevector(lo, hi, 0,1,2,3,4,5,6,7,8,9,10,11,12,13,14,15);
}

// 16B global -> LDS copy; async (ASYNCcnt-tracked, no VGPR round trip) when available.
static __device__ __forceinline__ void copy16_g2l(const __bf16* g, __bf16* l) {
#if HAVE_ASYNC_LDS
  __builtin_amdgcn_global_load_async_to_lds_b128(
      (GLOBAL_AS v4i*)(unsigned long long)g,
      (LDS_AS v4i*)l, 0, 0);
#else
  *(v8bf*)l = *(const v8bf*)g;
#endif
}

static __device__ __forceinline__ void wait_async0() {
#if HAVE_ASYNC_LDS
#if __has_builtin(__builtin_amdgcn_s_wait_asynccnt)
  __builtin_amdgcn_s_wait_asynccnt(0);
#else
  asm volatile("s_wait_asynccnt 0x0" ::: "memory");
#endif
#endif
}

// ---------------------------------------------------------------------------
// Small-M GEMM (recurrent steps): one wave = 16x64 tile, operands straight
// from L2 (weights are L2-resident across the 64 sequential steps).
// flags: bit0 = store bf16, bit1 = tanh, bit2 = permute row r -> (r%pB)*pT + r/pB
// ---------------------------------------------------------------------------
__global__ __launch_bounds__(32) void gemm_bf16_wmma(
    const __hip_bfloat16* __restrict__ A, const __hip_bfloat16* __restrict__ Bw,
    void* __restrict__ Cout, const float* __restrict__ addend,
    const float* __restrict__ bias,
    int K, long lda, long ldb, long ldc, int flags, int pB, int pT)
{
  const int lane = threadIdx.x & 31;
  const int l15  = lane & 15;
  const int half = lane >> 4;
  const long n0 = (long)blockIdx.x * 64;
  const long m0 = (long)blockIdx.y * 16;

  const __bf16* Ab = reinterpret_cast<const __bf16*>(A);
  const __bf16* Bb = reinterpret_cast<const __bf16*>(Bw);
  const __bf16* arow = Ab + (m0 + l15) * lda;
  const __bf16* brow[4];
#pragma unroll
  for (int j = 0; j < 4; ++j) brow[j] = Bb + (n0 + j * 16 + l15) * ldb;

  v8f acc[4] = {};
  for (int k0 = 0; k0 < K; k0 += 32) {
    const int kk = k0 + half * 8;  // ISA 16-bit frag layout: halves {0..7,16..23}/{8..15,24..31}
    __builtin_prefetch(arow + kk + 64, 0, 1);
    v16bf af = cat8(*(const v8bf*)(arow + kk), *(const v8bf*)(arow + kk + 16));
#pragma unroll
    for (int j = 0; j < 4; ++j) {
      v16bf bfr = cat8(*(const v8bf*)(brow[j] + kk), *(const v8bf*)(brow[j] + kk + 16));
      acc[j] = __builtin_amdgcn_wmma_f32_16x16x32_bf16(
          false, af, false, bfr, (short)0, acc[j], false, false);
    }
  }

#pragma unroll
  for (int r = 0; r < 8; ++r) {
    long row  = m0 + r + half * 8;
    long orow = (flags & 4) ? ((row % pB) * (long)pT + row / pB) : row;
#pragma unroll
    for (int j = 0; j < 4; ++j) {
      long col = n0 + j * 16 + l15;
      float v = acc[j][r];
      if (bias)   v += bias[col];
      if (addend) v += addend[row * ldc + col];
      if (flags & 2) v = tanhf(v);
      if (flags & 1) ((__hip_bfloat16*)Cout)[orow * ldc + col] = __float2bfloat16(v);
      else           ((float*)Cout)[orow * ldc + col] = v;
    }
  }
}

// ---------------------------------------------------------------------------
// Large GEMM: 8 waves / 256 threads, 64x128 block tile, Kc=64, double-buffered
// LDS (padded stride 72 to spread banks), async global->LDS staging overlapped
// with WMMA on the other buffer. B tile shared by 4 waves, A tile by 2 waves.
// ---------------------------------------------------------------------------
#define KC  64
#define LDP 72

__global__ __launch_bounds__(256) void gemm_bf16_wmma_tiled(
    const __hip_bfloat16* __restrict__ A, const __hip_bfloat16* __restrict__ Bw,
    void* __restrict__ Cout, const float* __restrict__ bias,
    int K, long lda, long ldb, long ldc, int flags, int pB, int pT)
{
  __shared__ __bf16 sA[2][64 * LDP];    // 18 KB
  __shared__ __bf16 sB[2][128 * LDP];   // 36 KB

  const int tid  = threadIdx.x;
  const int lane = tid & 31;
  const int l15  = lane & 15;
  const int half = lane >> 4;
  const int w    = tid >> 5;
  const int m_off = (w >> 1) * 16;      // 4 M sub-tiles
  const int n_off = (w & 1) * 64;       // 2 N sub-tiles (x4 frags)
  const long mbase = (long)blockIdx.y * 64;
  const long nbase = (long)blockIdx.x * 128;

  const __bf16* Ab = reinterpret_cast<const __bf16*>(A)  + mbase * lda;
  const __bf16* Bb = reinterpret_cast<const __bf16*>(Bw) + nbase * ldb;

  auto stage = [&](int buf, int k0) {
#pragma unroll
    for (int c = 0; c < 2; ++c) {           // A: 64 rows x 64 cols = 512 16B chunks
      int chunk = tid + c * 256;
      int row = chunk >> 3, col = (chunk & 7) * 8;
      copy16_g2l(Ab + (long)row * lda + k0 + col, &sA[buf][row * LDP + col]);
    }
#pragma unroll
    for (int c = 0; c < 4; ++c) {           // B: 128 rows x 64 cols = 1024 chunks
      int chunk = tid + c * 256;
      int row = chunk >> 3, col = (chunk & 7) * 8;
      copy16_g2l(Bb + (long)row * ldb + k0 + col, &sB[buf][row * LDP + col]);
    }
  };

  stage(0, 0);
  wait_async0();
  __syncthreads();

  v8f acc[4] = {};
  int cur = 0;
  for (int k0 = 0; k0 < K; k0 += KC) {
    if (k0 + KC < K) stage(cur ^ 1, k0 + KC);   // prefetch next chunk into other buffer
#pragma unroll
    for (int ks = 0; ks < KC; ks += 32) {
      const int kk = ks + half * 8;
      const __bf16* pa = &sA[cur][(m_off + l15) * LDP + kk];
      v16bf af = cat8(*(const v8bf*)pa, *(const v8bf*)(pa + 16));
#pragma unroll
      for (int j = 0; j < 4; ++j) {
        const __bf16* pb = &sB[cur][(n_off + j * 16 + l15) * LDP + kk];
        v16bf bfr = cat8(*(const v8bf*)pb, *(const v8bf*)(pb + 16));
        acc[j] = __builtin_amdgcn_wmma_f32_16x16x32_bf16(
            false, af, false, bfr, (short)0, acc[j], false, false);
      }
    }
    wait_async0();     // this wave's next-buffer loads landed
    __syncthreads();   // every wave done reading cur / writing cur^1
    cur ^= 1;
  }

#pragma unroll
  for (int r = 0; r < 8; ++r) {
    long row  = mbase + m_off + r + half * 8;
    long orow = (flags & 4) ? ((row % pB) * (long)pT + row / pB) : row;
#pragma unroll
    for (int j = 0; j < 4; ++j) {
      long col = nbase + n_off + j * 16 + l15;
      float v = acc[j][r];
      if (bias)   v += bias[col];
      if (flags & 2) v = tanhf(v);
      if (flags & 1) ((__hip_bfloat16*)Cout)[orow * ldc + col] = __float2bfloat16(v);
      else           ((float*)Cout)[orow * ldc + col] = v;
    }
  }
}

__global__ void f32_to_bf16(const float* __restrict__ s, __hip_bfloat16* __restrict__ d, long n) {
  long i = (long)blockIdx.x * blockDim.x + threadIdx.x;
  long stride = (long)gridDim.x * blockDim.x;
  for (; i < n; i += stride) d[i] = __float2bfloat16(s[i]);
}

__global__ void vec_add_f32(const float* __restrict__ a, const float* __restrict__ b,
                            float* __restrict__ o, int n) {
  int i = blockIdx.x * blockDim.x + threadIdx.x;
  if (i < n) o[i] = a[i] + b[i];
}

__global__ void fill_zero_u32(unsigned* __restrict__ p, long n) {
  long i = (long)blockIdx.x * blockDim.x + threadIdx.x;
  long stride = (long)gridDim.x * blockDim.x;
  for (; i < n; i += stride) p[i] = 0u;
}

// out[(s*B+b)*E + e] = bf16(emb[tok[b*S+s]][e])   (time-major rows for the big Gx GEMM)
__global__ void gather_embed_bf16(const int* __restrict__ tok, const float* __restrict__ emb,
                                  __hip_bfloat16* __restrict__ out, int B, int S, int E) {
  int row = blockIdx.x;
  int s = row / B, b = row % B;
  long id = tok[(long)b * S + s];
  const float* src = emb + id * (long)E;
  __hip_bfloat16* dst = out + (long)row * E;
  for (int e = threadIdx.x; e < E; e += blockDim.x) dst[e] = __float2bfloat16(src[e]);
}

// gates G[b,0:4H] (i,f,g,o) -> update c,h; mirror h to bf16 (strided); optional applied += h
__global__ void lstm_pointwise(const float* __restrict__ G, float* __restrict__ c,
                               float* __restrict__ h, __hip_bfloat16* __restrict__ hbf, int hbf_ld,
                               float* __restrict__ applied, int applied_ld, int B, int H) {
  int i = blockIdx.x * blockDim.x + threadIdx.x;
  if (i >= B * H) return;
  int b = i / H, k = i % H;
  const float* g = G + (long)b * 4 * H;
  float ig = 1.f / (1.f + expf(-g[k]));
  float fg = 1.f / (1.f + expf(-g[H + k]));
  float gg = tanhf(g[2 * H + k]);
  float og = 1.f / (1.f + expf(-g[3 * H + k]));
  float cn = fg * c[i] + ig * gg;
  c[i] = cn;
  float hn = og * tanhf(cn);
  h[i] = hn;
  hbf[(long)b * hbf_ld + k] = __float2bfloat16(hn);
  if (applied) applied[(long)b * applied_ld + k] += hn;
}

// h_d=hf+hb, c_d=cf+cb; concat[:,0:H]=bf16(h_d); concat[:,H:3H]=bf16(applied).
// (reference softmaxes a singleton axis => weights==1 => applied = sum_s enc_out, constant)
__global__ void combine_enc(const float* __restrict__ hf, const float* __restrict__ hb,
                            const float* __restrict__ cf, const float* __restrict__ cb,
                            float* __restrict__ hd, float* __restrict__ cd,
                            __hip_bfloat16* __restrict__ concat, int concat_ld,
                            const float* __restrict__ applied, int B, int H) {
  int i = blockIdx.x * blockDim.x + threadIdx.x;
  if (i < B * H) {
    float hv = hf[i] + hb[i];
    hd[i] = hv;
    cd[i] = cf[i] + cb[i];
    int b = i / H, k = i % H;
    concat[(long)b * concat_ld + k] = __float2bfloat16(hv);
  }
  if (i < B * 2 * H) {
    int b = i / (2 * H), k = i % (2 * H);
    concat[(long)b * concat_ld + H + k] = __float2bfloat16(applied[i]);
  }
}

__global__ __launch_bounds__(256) void logsoftmax_argmax(float* __restrict__ data,
                                                         int* __restrict__ preds, int V) {
  long row = blockIdx.x;
  float* p = data + row * (long)V;
  __shared__ float sv[256];
  __shared__ int si[256];
  int tid = threadIdx.x;
  float mx = -INFINITY; int mi = V;
  for (int i = tid; i < V; i += 256) {
    float v = p[i];
    if (v > mx || (v == mx && i < mi)) { mx = v; mi = i; }
  }
  sv[tid] = mx; si[tid] = mi; __syncthreads();
  for (int s = 128; s > 0; s >>= 1) {
    if (tid < s && (sv[tid + s] > sv[tid] || (sv[tid + s] == sv[tid] && si[tid + s] < si[tid]))) {
      sv[tid] = sv[tid + s]; si[tid] = si[tid + s];
    }
    __syncthreads();
  }
  float rmax = sv[0]; int rarg = si[0];
  __syncthreads();
  float sum = 0.f;
  for (int i = tid; i < V; i += 256) sum += expf(p[i] - rmax);
  sv[tid] = sum; __syncthreads();
  for (int s = 128; s > 0; s >>= 1) { if (tid < s) sv[tid] += sv[tid + s]; __syncthreads(); }
  float lse = rmax + logf(sv[0]);
  __syncthreads();
  for (int i = tid; i < V; i += 256) p[i] -= lse;
  if (tid == 0) preds[row] = rarg;
}

extern "C" void kernel_launch(void* const* d_in, const int* in_sizes, int n_in,
                              void* d_out, int out_size, void* d_ws, size_t ws_size,
                              hipStream_t stream) {
  (void)in_sizes; (void)n_in; (void)out_size; (void)ws_size;
  const int B = 32, S = 64, T = 64, E = 512, H = 1024, V = 32000, H4 = 4 * H;

  const int*   inputs  = (const int*)d_in[0];
  const int*   targets = (const int*)d_in[1];
  const float* emb_src = (const float*)d_in[2];
  const float* Wih_f = (const float*)d_in[3],  *Whh_f = (const float*)d_in[4];
  const float* bih_f = (const float*)d_in[5],  *bhh_f = (const float*)d_in[6];
  const float* Wih_b = (const float*)d_in[7],  *Whh_b = (const float*)d_in[8];
  const float* bih_b = (const float*)d_in[9],  *bhh_b = (const float*)d_in[10];
  const float* emb_tgt = (const float*)d_in[11];
  const float* Wih_d = (const float*)d_in[12], *Whh_d = (const float*)d_in[13];
  const float* bih_d = (const float*)d_in[14], *bhh_d = (const float*)d_in[15];
  /* d_in[16] = Wattn: dead compute (reference softmaxes a singleton axis) */
  const float* Wc   = (const float*)d_in[17];
  const float* bc   = (const float*)d_in[18];
  const float* Wout = (const float*)d_in[19];
  const float* bout = (const float*)d_in[20];

  char* ws = (char*)d_ws;
  size_t off = 0;
  auto alloc = [&](size_t bytes) -> void* {
    void* p = ws + off;
    off = (off + bytes + 255) & ~(size_t)255;
    return p;
  };

  __hip_bfloat16* WihF = (__hip_bfloat16*)alloc((size_t)H4 * E * 2);
  __hip_bfloat16* WhhF = (__hip_bfloat16*)alloc((size_t)H4 * H * 2);
  __hip_bfloat16* WihB = (__hip_bfloat16*)alloc((size_t)H4 * E * 2);
  __hip_bfloat16* WhhB = (__hip_bfloat16*)alloc((size_t)H4 * H * 2);
  __hip_bfloat16* WihD = (__hip_bfloat16*)alloc((size_t)H4 * E * 2);
  __hip_bfloat16* WhhD = (__hip_bfloat16*)alloc((size_t)H4 * H * 2);
  __hip_bfloat16* WcB  = (__hip_bfloat16*)alloc((size_t)H * 3 * H * 2);
  __hip_bfloat16* WoutB= (__hip_bfloat16*)alloc((size_t)V * H * 2);
  __hip_bfloat16* Asrc = (__hip_bfloat16*)alloc((size_t)S * B * E * 2);
  __hip_bfloat16* Atgt = (__hip_bfloat16*)alloc((size_t)T * B * E * 2);
  float* GxF   = (float*)alloc((size_t)S * B * H4 * 4);
  float* GxB   = (float*)alloc((size_t)S * B * H4 * 4);
  float* GxD   = (float*)alloc((size_t)T * B * H4 * 4);
  float* biasF = (float*)alloc((size_t)H4 * 4);
  float* biasB = (float*)alloc((size_t)H4 * 4);
  float* biasD = (float*)alloc((size_t)H4 * 4);
  float* h_f = (float*)alloc((size_t)B * H * 4);
  float* c_f = (float*)alloc((size_t)B * H * 4);
  float* h_b = (float*)alloc((size_t)B * H * 4);
  float* c_b = (float*)alloc((size_t)B * H * 4);
  float* h_d = (float*)alloc((size_t)B * H * 4);
  float* c_d = (float*)alloc((size_t)B * H * 4);
  __hip_bfloat16* hfbf = (__hip_bfloat16*)alloc((size_t)B * H * 2);
  __hip_bfloat16* hbbf = (__hip_bfloat16*)alloc((size_t)B * H * 2);
  float* applied = (float*)alloc((size_t)B * 2 * H * 4);
  __hip_bfloat16* concat = (__hip_bfloat16*)alloc((size_t)B * 3 * H * 2);
  float* G = (float*)alloc((size_t)B * H4 * 4);
  __hip_bfloat16* outsB = (__hip_bfloat16*)alloc((size_t)T * B * H * 2);

  auto cvt = [&](const float* s, __hip_bfloat16* d, long n) {
    long blk = (n + 255) / 256; if (blk > 8192) blk = 8192;
    f32_to_bf16<<<dim3((unsigned)blk), 256, 0, stream>>>(s, d, n);
  };
  auto zero = [&](void* p, long bytes) {
    long n = bytes / 4;
    long blk = (n + 255) / 256; if (blk > 4096) blk = 4096;
    fill_zero_u32<<<dim3((unsigned)blk), 256, 0, stream>>>((unsigned*)p, n);
  };
  auto gemm = [&](const __hip_bfloat16* A, const __hip_bfloat16* Bw, void* C,
                  const float* add, const float* bias, int M, int N, int K,
                  long lda, long ldb, long ldc, int flags) {
    dim3 grid((unsigned)(N / 64), (unsigned)(M / 16));
    gemm_bf16_wmma<<<grid, 32, 0, stream>>>(A, Bw, C, add, bias, K, lda, ldb, ldc, flags, B, T);
  };
  auto gemmT = [&](const __hip_bfloat16* A, const __hip_bfloat16* Bw, void* C,
                   const float* bias, int M, int N, int K,
                   long lda, long ldb, long ldc, int flags) {
    dim3 grid((unsigned)(N / 128), (unsigned)(M / 64));
    gemm_bf16_wmma_tiled<<<grid, 256, 0, stream>>>(A, Bw, C, bias, K, lda, ldb, ldc, flags, B, T);
  };

  // --- one-time weight conversion to bf16 (reused 64x for each Whh / once for Wout) ---
  cvt(Wih_f, WihF, (long)H4 * E);  cvt(Whh_f, WhhF, (long)H4 * H);
  cvt(Wih_b, WihB, (long)H4 * E);  cvt(Whh_b, WhhB, (long)H4 * H);
  cvt(Wih_d, WihD, (long)H4 * E);  cvt(Whh_d, WhhD, (long)H4 * H);
  cvt(Wc,    WcB,  (long)H * 3 * H);
  cvt(Wout,  WoutB,(long)V * H);

  vec_add_f32<<<(H4 + 255) / 256, 256, 0, stream>>>(bih_f, bhh_f, biasF, H4);
  vec_add_f32<<<(H4 + 255) / 256, 256, 0, stream>>>(bih_b, bhh_b, biasB, H4);
  vec_add_f32<<<(H4 + 255) / 256, 256, 0, stream>>>(bih_d, bhh_d, biasD, H4);

  // --- embedding gathers (time-major rows: s*B+b) ---
  gather_embed_bf16<<<S * B, 256, 0, stream>>>(inputs,  emb_src, Asrc, B, S, E);
  gather_embed_bf16<<<T * B, 256, 0, stream>>>(targets, emb_tgt, Atgt, B, T, E);

  // --- hoisted input projections for all timesteps (LDS-tiled WMMA GEMMs) ---
  gemmT(Asrc, WihF, GxF, nullptr, S * B, H4, E, E, E, H4, 0);
  gemmT(Asrc, WihB, GxB, nullptr, S * B, H4, E, E, E, H4, 0);
  gemmT(Atgt, WihD, GxD, nullptr, T * B, H4, E, E, E, H4, 0);

  // --- init recurrent state ---
  zero(h_f, (long)B * H * 4); zero(c_f, (long)B * H * 4);
  zero(h_b, (long)B * H * 4); zero(c_b, (long)B * H * 4);
  zero(hfbf, (long)B * H * 2); zero(hbbf, (long)B * H * 2);
  zero(applied, (long)B * 2 * H * 4);

  const int pwBlocks = (B * H + 255) / 256;

  // --- forward encoder LSTM (sequential; only h@Whh^T remains per step) ---
  for (int t = 0; t < S; ++t) {
    gemm(hfbf, WhhF, G, GxF + (long)t * B * H4, biasF, B, H4, H, H, H, H4, 0);
    lstm_pointwise<<<pwBlocks, 256, 0, stream>>>(G, c_f, h_f, hfbf, H, applied, 2 * H, B, H);
  }
  // --- backward encoder LSTM (reversed input = reversed Gx row blocks) ---
  for (int t = 0; t < S; ++t) {
    gemm(hbbf, WhhB, G, GxB + (long)(S - 1 - t) * B * H4, biasB, B, H4, H, H, H, H4, 0);
    lstm_pointwise<<<pwBlocks, 256, 0, stream>>>(G, c_b, h_b, hbbf, H, applied + H, 2 * H, B, H);
  }

  // --- combine encoder states; build constant concat[:,H:3H] = sum_s enc_out ---
  combine_enc<<<(B * 2 * H + 255) / 256, 256, 0, stream>>>(
      h_f, h_b, c_f, c_b, h_d, c_d, concat, 3 * H, applied, B, H);

  // --- decoder: gates GEMM -> pointwise -> fused tanh(concat @ Wc^T + bc) into bf16 outs ---
  for (int t = 0; t < T; ++t) {
    gemm(concat, WhhD, G, GxD + (long)t * B * H4, biasD, B, H4, H, 3 * H, H, H4, 0);
    lstm_pointwise<<<pwBlocks, 256, 0, stream>>>(G, c_d, h_d, concat, 3 * H, nullptr, 0, B, H);
    gemm(concat, WcB, outsB + (long)t * B * H, nullptr, bc, B, H, 3 * H, 3 * H, 3 * H, H,
         /*bf16 out + tanh*/ 1 | 2);
  }

  // --- dominant GEMM: logits = outs @ Wout^T + bout -> d_out, with (t*B+b)->(b*T+t) row
  //     permutation fused into the epilogue (no 262MB scratch pass) ---
  gemmT(outsB, WoutB, d_out, bout, T * B, V, H, H, H, V, /*permute*/ 4);

  // --- in-place log_softmax over V + argmax; preds appended after the scores block ---
  logsoftmax_argmax<<<B * T, 256, 0, stream>>>((float*)d_out, (int*)d_out + (size_t)B * T * V, V);
}
